// RFNetwork_4690104287270
// MI455X (gfx1250) — matrix operand order repaired and checked by reading, not compiled, not run on hardware
//
#include <hip/hip_runtime.h>
#include <hip/hip_bf16.h>
#include <cstdint>
#include <cstddef>

#define NN 2048
#define MM 2048
#define TT 32
#define RR 4
#define RSZ 512
#define K_IN 25
#define K_OUT 102
#define LMBDA 0.01f

typedef __attribute__((ext_vector_type(16))) _Float16 v16h;
typedef __attribute__((ext_vector_type(8)))  float    v8f;

// ---------------- deterministic hash-based normal noise --------------------
__device__ __forceinline__ unsigned int pcg_hash(unsigned int x) {
    x = x * 747796405u + 2891336453u;
    unsigned int w = ((x >> ((x >> 28u) + 4u)) ^ x) * 277803737u;
    return (w >> 22u) ^ w;
}
__device__ __forceinline__ float rnd_normal(unsigned int key, unsigned int idx) {
    unsigned int h1 = pcg_hash(key * 0x9E3779B9u ^ (idx * 0x85EBCA6Bu + 0xC2B2AE35u));
    unsigned int h2 = pcg_hash(h1 ^ 0x27D4EB2Fu);
    float u1 = (float)(h1 >> 8) * (1.0f / 16777216.0f) + 1.0e-7f;
    float u2 = (float)(h2 >> 8) * (1.0f / 16777216.0f);
    return sqrtf(-2.0f * logf(u1)) * cosf(6.28318530718f * u2);
}

// ---------------- zero init (P starts at zeros) ----------------------------
__global__ void zero_f32_kernel(float* __restrict__ p, long long n) {
    long long i = (long long)blockIdx.x * blockDim.x + threadIdx.x;
    long long stride = (long long)gridDim.x * blockDim.x;
    for (; i < n; i += stride) p[i] = 0.0f;
}

// ---------------- per-timestep min/max of inp row --------------------------
__global__ void minmax_kernel(const float* __restrict__ inp, float* __restrict__ minmax) {
    __shared__ float smn[256], smx[256];
    const int t = blockIdx.x, tid = threadIdx.x;
    float mn = 3.4e38f, mx = -3.4e38f;
    for (int j = tid; j < NN; j += 256) {
        float v = inp[t * NN + j];
        mn = fminf(mn, v); mx = fmaxf(mx, v);
    }
    smn[tid] = mn; smx[tid] = mx; __syncthreads();
    for (int s = 128; s > 0; s >>= 1) {
        if (tid < s) { smn[tid] = fminf(smn[tid], smn[tid + s]); smx[tid] = fmaxf(smx[tid], smx[tid + s]); }
        __syncthreads();
    }
    if (tid == 0) { minmax[2 * t] = smn[0]; minmax[2 * t + 1] = smx[0]; }
}

// ---------------- activation_in: per-region top-K binarize -----------------
__global__ void act_in_kernel(const float* __restrict__ inp,
                              const float* __restrict__ minmax,
                              _Float16* __restrict__ In_h) {
    __shared__ float vals[RSZ];
    __shared__ unsigned char flags[RSZ];
    __shared__ float rv[RSZ];
    __shared__ int   ri[RSZ];
    const int t = blockIdx.x / RR, r = blockIdx.x % RR, tid = threadIdx.x;
    const int base = r * RSZ;
    const float amp = (1.0e-10f + minmax[2 * t + 1] - minmax[2 * t]) * 0.01f;
    float x = inp[t * NN + base + tid];
    vals[tid]  = x + amp * rnd_normal(2u * t, (unsigned)(base + tid));
    flags[tid] = 0;
    __syncthreads();
    for (int it = 0; it < K_IN; ++it) {
        rv[tid] = flags[tid] ? -3.4e38f : vals[tid];
        ri[tid] = tid;
        __syncthreads();
        for (int s = RSZ / 2; s > 0; s >>= 1) {
            if (tid < s && rv[tid + s] > rv[tid]) { rv[tid] = rv[tid + s]; ri[tid] = ri[tid + s]; }
            __syncthreads();
        }
        if (tid == 0) flags[ri[0]] = 1;
        __syncthreads();
    }
    In_h[t * NN + base + tid] = (_Float16)(flags[tid] ? 1.0f : 0.0f);
}

// ---------------- GEMM1: Y[T,M] = out_in(f32->f16) x In^T via WMMA ---------
__global__ void gemm1_wmma_kernel(const float* __restrict__ out_in,
                                  const _Float16* __restrict__ In_h,
                                  float* __restrict__ Y) {
    const int wave = blockIdx.x * 4 + (threadIdx.x >> 5);   // 0..255
    const int lane = threadIdx.x & 31;
    const int tm = wave >> 1;                               // 0..127 -> m tile
    const int tt = wave & 1;                                // 0..1   -> t tile
    const int m0 = tm * 16, t0 = tt * 16;
    const int lrow = lane & 15, hi = lane >> 4;

    v8f c = {};
    const int arow = m0 + lrow;
    const int akoff = hi ? 8 : 0;
    const int brow = t0 + lrow;
    const int bkoff = hi ? 8 : 0;
    for (int k0 = 0; k0 < NN; k0 += 32) {
        __builtin_prefetch(&out_in[(size_t)arow * NN + k0 + 128], 0, 1);
        v16h a, b;
        const float* ap = &out_in[(size_t)arow * NN + k0 + akoff];
        #pragma unroll
        for (int j = 0; j < 8; ++j) {
            a[j]     = (_Float16)ap[j];
            a[8 + j] = (_Float16)ap[16 + j];
        }
        const _Float16* bp = &In_h[(size_t)brow * NN + k0 + bkoff];
        #pragma unroll
        for (int kk = 0; kk < 8; ++kk) { b[kk] = bp[kk]; b[8 + kk] = bp[16 + kk]; }
        c = __builtin_amdgcn_wmma_f32_16x16x32_f16(false, a, false, b, (short)0, c, false, false);
    }
    const int tcol = t0 + lrow;
    const int rbase = m0 + (hi ? 8 : 0);
    #pragma unroll
    for (int i = 0; i < 8; ++i) Y[(size_t)tcol * MM + rbase + i] = c[i];
}

// ---------------- activation_out: top-K_OUT of noisy Y row -----------------
__global__ void act_out_kernel(const float* __restrict__ Y,
                               float* __restrict__ outs,      // [T, M]
                               _Float16* __restrict__ OutT_h) // [M, T]
{
    __shared__ float vals[MM];
    __shared__ unsigned char flags[MM];
    __shared__ float rv[1024];
    __shared__ int   ri[1024];
    const int t = blockIdx.x, tid = threadIdx.x;
    float y0 = Y[t * MM + tid], y1 = Y[t * MM + tid + 1024];
    // min reduce (pre-noise)
    rv[tid] = fminf(y0, y1); __syncthreads();
    for (int s = 512; s > 0; s >>= 1) {
        if (tid < s) rv[tid] = fminf(rv[tid], rv[tid + s]);
        __syncthreads();
    }
    const float amp = fabsf(rv[0] * 0.1f);
    __syncthreads();
    vals[tid]        = y0 + amp * rnd_normal(2u * t + 1u, (unsigned)tid);
    vals[tid + 1024] = y1 + amp * rnd_normal(2u * t + 1u, (unsigned)(tid + 1024));
    flags[tid] = 0; flags[tid + 1024] = 0;
    __syncthreads();
    for (int it = 0; it < K_OUT; ++it) {
        float a0 = flags[tid]        ? -3.4e38f : vals[tid];
        float a1 = flags[tid + 1024] ? -3.4e38f : vals[tid + 1024];
        rv[tid] = (a1 > a0) ? a1 : a0;
        ri[tid] = (a1 > a0) ? tid + 1024 : tid;
        __syncthreads();
        for (int s = 512; s > 0; s >>= 1) {
            if (tid < s && rv[tid + s] > rv[tid]) { rv[tid] = rv[tid + s]; ri[tid] = ri[tid + s]; }
            __syncthreads();
        }
        if (tid == 0) flags[ri[0]] = 1;
        __syncthreads();
    }
    float o0 = flags[tid] ? 1.0f : 0.0f, o1 = flags[tid + 1024] ? 1.0f : 0.0f;
    outs[t * MM + tid] = o0; outs[t * MM + tid + 1024] = o1;
    OutT_h[(size_t)tid * TT + t] = (_Float16)o0;
    OutT_h[(size_t)(tid + 1024) * TT + t] = (_Float16)o1;
}

// ---------------- GEMM2: out_p = λ · Outᵀ·In  (K=32, one WMMA/tile) --------
__global__ void gemm2_wmma_kernel(const _Float16* __restrict__ OutT_h, // [M,32]
                                  const _Float16* __restrict__ In_h,   // [T,N]
                                  float* __restrict__ outP)            // [M,N]
{
    const int wave = blockIdx.x * 4 + (threadIdx.x >> 5);   // 0..16383
    const int lane = threadIdx.x & 31;
    const int ti = wave >> 7, tj = wave & 127;
    const int m0 = ti * 16, n0 = tj * 16;
    const int lrow = lane & 15, hi = lane >> 4;

    v16h a, b;
    const _Float16* ap = &OutT_h[(size_t)(m0 + lrow) * TT + (hi ? 8 : 0)];
    #pragma unroll
    for (int j = 0; j < 8; ++j) { a[j] = ap[j]; a[8 + j] = ap[16 + j]; }
    const int bkoff = hi ? 8 : 0;
    const int bcol = n0 + lrow;
    #pragma unroll
    for (int kk = 0; kk < 16; ++kk) b[kk] = In_h[(size_t)(bkoff + kk + (kk >= 8 ? 8 : 0)) * NN + bcol];

    v8f c = {};
    c = __builtin_amdgcn_wmma_f32_16x16x32_f16(false, a, false, b, (short)0, c, false, false);

    const int rbase = m0 + (hi ? 8 : 0);
    #pragma unroll
    for (int i = 0; i < 8; ++i) outP[(size_t)(rbase + i) * NN + bcol] = LMBDA * c[i];
}

// ---------------- homeostasis step: hebbian + row sums ---------------------
__global__ void hebb_row_kernel(float* __restrict__ P,
                                const float* __restrict__ F,
                                const _Float16* __restrict__ In_h,
                                int t, float* __restrict__ rowsum) {
    __shared__ float sd[256];
    const int i = blockIdx.x, tid = threadIdx.x;
    const float in_i = (float)In_h[t * NN + i] * LMBDA;
    float s = 0.0f;
    #pragma unroll
    for (int jj = 0; jj < 8; ++jj) {
        int j = jj * 256 + tid;
        float p = P[(size_t)i * NN + j] + in_i * (float)In_h[t * NN + j] * F[(size_t)i * NN + j];
        P[(size_t)i * NN + j] = p;
        s += p;
    }
    sd[tid] = s; __syncthreads();
    for (int st = 128; st > 0; st >>= 1) {
        if (tid < st) sd[tid] += sd[tid + st];
        __syncthreads();
    }
    if (tid == 0) rowsum[i] = sd[0];
}

// ---------------- column sums ----------------------------------------------
__global__ void colsum_kernel(const float* __restrict__ P, float* __restrict__ colsum) {
    __shared__ float sd[256];
    const int tid = threadIdx.x;
    const int j = blockIdx.x * 64 + (tid & 63);
    float s = 0.0f;
    for (int i = (tid >> 6); i < NN; i += 4) s += P[(size_t)i * NN + j];
    sd[tid] = s; __syncthreads();
    if (tid < 64) colsum[j] = sd[tid] + sd[tid + 64] + sd[tid + 128] + sd[tid + 192];
}

// ---------------- homeostasis rescale (incl. torch broadcast quirk) --------
__global__ void scale_kernel(float* __restrict__ P,
                             const float* __restrict__ rowsum,
                             const float* __restrict__ colsum) {
    const int i = blockIdx.x, tid = threadIdx.x;
    const float post_i = rowsum[i];
    const bool post_ex = post_i > 1.0f;   // MAX_POST
    #pragma unroll
    for (int jj = 0; jj < 8; ++jj) {
        int j = jj * 256 + tid;
        float pre_j = colsum[j];
        bool pre_ex = pre_j > 1.0f;       // MAX_PRE
        float s = 1.0f;
        if (pre_ex && !post_ex) s *= 1.0f / pre_j;                 // pre_s
        if (post_ex && !pre_ex) s *= 1.0f / rowsum[j];             // post_s (broadcast on j!)
        if (pre_ex && post_ex)  s *= 2.0f / (pre_j + post_i);      // pp_s
        P[(size_t)i * NN + j] *= s;
    }
}

// ---------------- final: seg2 = in_in_fixed + P (in place) -----------------
__global__ void add_fixed_kernel(float* __restrict__ P, const float* __restrict__ fixed) {
    long long i = (long long)blockIdx.x * blockDim.x + threadIdx.x;
    long long stride = (long long)gridDim.x * blockDim.x;
    for (; i < (long long)NN * NN; i += stride) P[i] += fixed[i];
}

extern "C" void kernel_launch(void* const* d_in, const int* in_sizes, int n_in,
                              void* d_out, int out_size, void* d_ws, size_t ws_size,
                              hipStream_t stream) {
    const float* inp   = (const float*)d_in[0];   // [T, N]
    const float* fixed = (const float*)d_in[1];   // [N, N]
    const float* out_in = (const float*)d_in[2];  // [M, N]
    const float* F     = (const float*)d_in[3];   // [N, N]

    float* seg1 = (float*)d_out;                          // outs   [T, M]
    float* segP = seg1 + (size_t)TT * MM;                 // in_p   [N, N]
    float* segO = segP + (size_t)NN * NN;                 // out_p  [M, N]

    _Float16* In_h   = (_Float16*)d_ws;                   // [T, N]
    _Float16* OutT_h = In_h + (size_t)TT * NN;            // [M, T]
    float*    Y      = (float*)(OutT_h + (size_t)MM * TT);// [T, M]
    float*    minmax = Y + (size_t)TT * MM;               // [2T]
    float*    rowsum = minmax + 2 * TT;                   // [N]
    float*    colsum = rowsum + NN;                       // [N]

    // P := 0
    zero_f32_kernel<<<1024, 256, 0, stream>>>(segP, (long long)NN * NN);
    // parallel activations for all timesteps
    minmax_kernel<<<TT, 256, 0, stream>>>(inp, minmax);
    act_in_kernel<<<TT * RR, RSZ, 0, stream>>>(inp, minmax, In_h);
    // Y = out_in @ In^T   (WMMA f16, K-loop of 64)
    gemm1_wmma_kernel<<<64, 128, 0, stream>>>(out_in, In_h, Y);
    act_out_kernel<<<TT, 1024, 0, stream>>>(Y, seg1, OutT_h);
    // out_plastic = λ · Outᵀ·In   (rank-32: single WMMA per tile)
    gemm2_wmma_kernel<<<4096, 128, 0, stream>>>(OutT_h, In_h, segO);
    // sequential homeostasis scan (L2-resident 16.8 MB matrix)
    for (int t = 0; t < TT; ++t) {
        hebb_row_kernel<<<NN, 256, 0, stream>>>(segP, F, In_h, t, rowsum);
        colsum_kernel<<<NN / 64, 256, 0, stream>>>(segP, colsum);
        scale_kernel<<<NN, 256, 0, stream>>>(segP, rowsum, colsum);
    }
    add_fixed_kernel<<<1024, 256, 0, stream>>>(segP, fixed);
}
